// CausalAttention_18064632447487
// MI455X (gfx1250) — matrix operand order repaired and checked by reading, not compiled, not run on hardware
//
#include <hip/hip_runtime.h>
#include <hip/hip_bf16.h>

// ---------------------------------------------------------------------------
// CausalAttention for MI455X (gfx1250, wave32, WMMA f32_16x16x32_f16)
// B=2, N=2048, DIM=1024, HEADS=16, DIM_HEAD=64, INNER=1024
// ---------------------------------------------------------------------------

typedef _Float16 v8h  __attribute__((ext_vector_type(8)));
typedef _Float16 v16h __attribute__((ext_vector_type(16)));
typedef float    v8f  __attribute__((ext_vector_type(8)));
typedef float    v4f  __attribute__((ext_vector_type(4)));
typedef _Float16 v4h  __attribute__((ext_vector_type(4)));

#define SEQ   2048
#define DIMM  1024
#define NHEAD 16
#define DHEAD 64
#define BROWS 4096   // B*N

static __device__ __forceinline__ v16h ld_frag(const _Float16* p0, const _Float16* p1) {
    v8h lo = *(const v8h*)p0;
    v8h hi = *(const v8h*)p1;
    return __builtin_shufflevector(lo, hi, 0,1,2,3,4,5,6,7,8,9,10,11,12,13,14,15);
}

static __device__ __forceinline__ v8f wmma_f16(v16h a, v16h b, v8f c) {
    return __builtin_amdgcn_wmma_f32_16x16x32_f16(false, a, false, b, (short)0, c, false, false);
}

// ---------------------------------------------------------------------------
// LayerNorm (fp32) -> f16 output. One row of 1024 per block (256 threads).
// ---------------------------------------------------------------------------
__global__ __launch_bounds__(256) void ln_kernel(const float* __restrict__ x,
                                                 const float* __restrict__ w,
                                                 const float* __restrict__ b,
                                                 _Float16* __restrict__ xn) {
    const int row = blockIdx.x;
    const int tid = threadIdx.x;
    const float* xr = x + (size_t)row * DIMM;
    float s = 0.f, s2 = 0.f;
    v4f xv = *(const v4f*)(xr + tid * 4);
    #pragma unroll
    for (int i = 0; i < 4; ++i) { s += xv[i]; s2 += xv[i] * xv[i]; }
    __shared__ float sb[256], s2b[256];
    sb[tid] = s; s2b[tid] = s2; __syncthreads();
    for (int off = 128; off > 0; off >>= 1) {
        if (tid < off) { sb[tid] += sb[tid + off]; s2b[tid] += s2b[tid + off]; }
        __syncthreads();
    }
    const float mu  = sb[0] * (1.f / DIMM);
    const float var = s2b[0] * (1.f / DIMM) - mu * mu;
    const float rstd = rsqrtf(var + 1e-5f);
    v4f wv = *(const v4f*)(w + tid * 4);
    v4f bv = *(const v4f*)(b + tid * 4);
    v4h o;
    #pragma unroll
    for (int i = 0; i < 4; ++i) o[i] = (_Float16)((xv[i] - mu) * rstd * wv[i] + bv[i]);
    *(v4h*)(xn + (size_t)row * DIMM + tid * 4) = o;
}

// ---------------------------------------------------------------------------
// f32 -> f16 cast, 4 elements/thread (b128 load, b64 store)
// ---------------------------------------------------------------------------
__global__ __launch_bounds__(256) void cast_f16_kernel(const float* __restrict__ src,
                                                       _Float16* __restrict__ dst, int n4) {
    int i = blockIdx.x * 256 + threadIdx.x;
    if (i < n4) {
        v4f v = *(const v4f*)(src + (size_t)i * 4);
        v4h o;
        #pragma unroll
        for (int k = 0; k < 4; ++k) o[k] = (_Float16)v[k];
        *(v4h*)(dst + (size_t)i * 4) = o;
    }
}

// ---------------------------------------------------------------------------
// WMMA GEMM: C[M x Nd] = A[M x K] * Bm[K x Nd], A/Bm f16 row-major.
// Block tile 128x128, BK=32, 256 threads = 8 waves (4 along M x 2 along N),
// each wave computes 32x64 = 2x4 WMMA tiles. Register double-buffered staging.
// MODE 0: epilogue splits columns into Q (scaled 0.125) / K / V head-major f16.
// MODE 1: epilogue writes fp32 to outf.
// ---------------------------------------------------------------------------
template <int MODE>
__global__ __launch_bounds__(256, 1) void gemm_wmma(
        const _Float16* __restrict__ A, const _Float16* __restrict__ Bm,
        int M, int Nd, int K,
        _Float16* __restrict__ qo, _Float16* __restrict__ ko,
        _Float16* __restrict__ vo, float* __restrict__ outf) {
    __shared__ __align__(16) _Float16 As[128 * 40];   // [m][k], stride 40 halves (80B)
    __shared__ __align__(16) _Float16 Bs[128 * 40];   // transposed: [n][k]

    const int tid  = threadIdx.x;
    const int lane = tid & 31;
    const int wv   = tid >> 5;
    const int hl   = lane & 15;
    const int grp  = lane >> 4;
    const int o1   = grp * 8;
    const int o2   = 16 + grp * 8;
    const int wm   = wv & 3;   // 0..3, 32 rows each
    const int wn   = wv >> 2;  // 0..1, 64 cols each
    const int m0   = blockIdx.y * 128;
    const int n0   = blockIdx.x * 128;

    // staging decomposition (per thread, 2 chunks each for A and B)
    const int ar0 = tid >> 2, ac0 = (tid & 3) * 8;          // A: idx = tid, tid+256
    const int ar1 = (tid + 256) >> 2, ac1 = ((tid + 256) & 3) * 8;
    const int bk0 = tid >> 4, bn0 = (tid & 15) * 8;         // B: idx = tid, tid+256
    const int bk1 = (tid + 256) >> 4, bn1 = ((tid + 256) & 15) * 8;

    v8h aR0, aR1, bR0, bR1;
    // preload k-tile 0 into registers
    aR0 = *(const v8h*)(A + (size_t)(m0 + ar0) * K + ac0);
    aR1 = *(const v8h*)(A + (size_t)(m0 + ar1) * K + ac1);
    bR0 = *(const v8h*)(Bm + (size_t)bk0 * Nd + n0 + bn0);
    bR1 = *(const v8h*)(Bm + (size_t)bk1 * Nd + n0 + bn1);
    // commit to LDS
    *(v8h*)&As[ar0 * 40 + ac0] = aR0;
    *(v8h*)&As[ar1 * 40 + ac1] = aR1;
    #pragma unroll
    for (int i = 0; i < 8; ++i) Bs[(bn0 + i) * 40 + bk0] = bR0[i];
    #pragma unroll
    for (int i = 0; i < 8; ++i) Bs[(bn1 + i) * 40 + bk1] = bR1[i];

    v8f acc[2][4] = {};

    for (int kt = 0; kt < K; kt += 32) {
        __syncthreads();
        const bool more = (kt + 32) < K;
        if (more) {  // issue next tile's global loads; latency hidden by compute
            const int kn = kt + 32;
            aR0 = *(const v8h*)(A + (size_t)(m0 + ar0) * K + kn + ac0);
            aR1 = *(const v8h*)(A + (size_t)(m0 + ar1) * K + kn + ac1);
            bR0 = *(const v8h*)(Bm + (size_t)(kn + bk0) * Nd + n0 + bn0);
            bR1 = *(const v8h*)(Bm + (size_t)(kn + bk1) * Nd + n0 + bn1);
        }

        v16h af[2], bf[4];
        #pragma unroll
        for (int i = 0; i < 2; ++i) {
            int r = wm * 32 + i * 16 + hl;
            af[i] = ld_frag(&As[r * 40 + o1], &As[r * 40 + o2]);
        }
        #pragma unroll
        for (int j = 0; j < 4; ++j) {
            int c = wn * 64 + j * 16 + hl;
            bf[j] = ld_frag(&Bs[c * 40 + o1], &Bs[c * 40 + o2]);
        }
        #pragma unroll
        for (int i = 0; i < 2; ++i)
            #pragma unroll
            for (int j = 0; j < 4; ++j)
                acc[i][j] = wmma_f16(af[i], bf[j], acc[i][j]);

        __syncthreads();
        if (more) {
            *(v8h*)&As[ar0 * 40 + ac0] = aR0;
            *(v8h*)&As[ar1 * 40 + ac1] = aR1;
            #pragma unroll
            for (int i = 0; i < 8; ++i) Bs[(bn0 + i) * 40 + bk0] = bR0[i];
            #pragma unroll
            for (int i = 0; i < 8; ++i) Bs[(bn1 + i) * 40 + bk1] = bR1[i];
        }
    }

    // epilogue: D layout -> lane hl = column, vgpr r = row (r + grp*8) in 16-tile
    #pragma unroll
    for (int i = 0; i < 2; ++i) {
        #pragma unroll
        for (int j = 0; j < 4; ++j) {
            const int grb = m0 + wm * 32 + i * 16 + grp * 8;
            const int gc  = n0 + wn * 64 + j * 16 + hl;
            #pragma unroll
            for (int r = 0; r < 8; ++r) {
                const int gr = grb + r;
                const float val = acc[i][j][r];
                if (MODE == 0) {
                    const int which = gc >> 10;          // 0=q 1=k 2=v
                    const int cc = gc & 1023;
                    const int h  = cc >> 6;
                    const int d  = cc & 63;
                    const int bb = gr >> 11;             // batch
                    const int nn = gr & 2047;            // seq pos
                    const size_t idx = (((size_t)(bb * NHEAD + h) * SEQ + nn) * DHEAD + d);
                    if (which == 0)      qo[idx] = (_Float16)(val * 0.125f);
                    else if (which == 1) ko[idx] = (_Float16)val;
                    else                 vo[idx] = (_Float16)val;
                } else {
                    outf[(size_t)gr * Nd + gc] = val;
                }
            }
        }
    }
}

// ---------------------------------------------------------------------------
// Causal flash attention. Grid: (B*H, N/128). Block: 256 threads = 8 waves,
// each wave owns 16 query rows. Key tiles of 32, K/V staged in LDS with
// register double-buffering; P transposed through LDS between WMMA stages.
// Wave-uniform (readfirstlane'd) branches skip masking on non-diagonal tiles
// and the whole compute body on fully-masked tiles -> scalar s_cbranch,
// EXEC stays all-ones around every WMMA.
// Q was pre-scaled by DIM_HEAD^-0.5 in the QKV epilogue.
// Output written head-interleaved: O[(b*N + n)*1024 + h*64 + d] (f16).
// ---------------------------------------------------------------------------
__global__ __launch_bounds__(256, 1) void attn_kernel(
        const _Float16* __restrict__ Q, const _Float16* __restrict__ Kb,
        const _Float16* __restrict__ Vb, _Float16* __restrict__ O) {
    const int bh = blockIdx.x;          // b*16 + h
    const int qb = blockIdx.y;          // 128-row query block
    const int tid  = threadIdx.x;
    const int lane = tid & 31;
    const int wv   = tid >> 5;          // 0..7
    const int hl   = lane & 15;
    const int grp  = lane >> 4;
    const int o1   = grp * 8;
    const int o2   = 16 + grp * 8;

    const _Float16* Qh = Q  + (size_t)bh * SEQ * DHEAD;
    const _Float16* Kh = Kb + (size_t)bh * SEQ * DHEAD;
    const _Float16* Vh = Vb + (size_t)bh * SEQ * DHEAD;

    const int q0 = qb * 128 + wv * 16;
    const int q0u = __builtin_amdgcn_readfirstlane(q0);  // provably wave-uniform
    const int qlane = q0 + grp * 8;     // row base for this lane's D-layout slots

    // Q A-fragments (16 x 64 -> two 16x32 fragments), straight from global
    v16h qf[2];
    #pragma unroll
    for (int c = 0; c < 2; ++c) {
        const _Float16* qr = Qh + (size_t)(q0 + hl) * DHEAD + c * 32;
        qf[c] = ld_frag(qr + o1, qr + o2);
    }

    __shared__ __align__(16) _Float16 Ks[32 * 72];       // [key][d], stride 72
    __shared__ __align__(16) _Float16 Vt[64 * 40];       // [d][key], stride 40
    __shared__ __align__(16) _Float16 Ps[8][16 * 40];    // per-wave P tile [m][k]

    // staging decomposition: K -> 1 chunk/thread, V -> 8 halves/thread (transpose)
    const int skr = tid >> 3, skc = (tid & 7) * 8;       // K: row, col8
    const int svk = tid & 31, svd = (tid >> 5) * 8;      // V: key, d8

    v8h kR, vR;
    kR = *(const v8h*)(Kh + (size_t)skr * DHEAD + skc);
    vR = *(const v8h*)(Vh + (size_t)svk * DHEAD + svd);
    *(v8h*)&Ks[skr * 72 + skc] = kR;
    #pragma unroll
    for (int i = 0; i < 8; ++i) Vt[(svd + i) * 40 + svk] = vR[i];

    v8f oacc[4] = {};
    float mrow[8], lrow[8];
    #pragma unroll
    for (int r = 0; r < 8; ++r) { mrow[r] = -3e38f; lrow[r] = 0.f; }

    const int nkt = qb * 4 + 4;   // key tiles covering keys 0 .. qb*128+127
    for (int t = 0; t < nkt; ++t) {
        const int kt = t * 32;
        __syncthreads();   // LDS tile t ready for all waves
        if (t + 1 < nkt) {  // register-prefetch tile t+1 (latency hidden by compute)
            const size_t nb = (size_t)(kt + 32) * DHEAD;
            kR = *(const v8h*)(Kh + nb + (size_t)skr * DHEAD + skc);
            vR = *(const v8h*)(Vh + nb + (size_t)svk * DHEAD + svd);
        }
        if (t + 2 < nkt) {  // L2 prefetch tile t+2 (global_prefetch_b8)
            const size_t pb = (size_t)(kt + 64) * DHEAD + tid * 16;
            __builtin_prefetch(Kh + pb, 0, 3);
            __builtin_prefetch(Vh + pb, 0, 3);
        }

        // skip tiles entirely above the diagonal for this wave (scalar branch)
        if (kt <= q0u + 15) {
            // S = Q @ K^T : 16 x 32, two 16-column tiles, accumulate over d (2x32)
            v8f s[2] = {};
            #pragma unroll
            for (int nt = 0; nt < 2; ++nt) {
                const int krow = nt * 16 + hl;
                #pragma unroll
                for (int c = 0; c < 2; ++c) {
                    v16h kf = ld_frag(&Ks[krow * 72 + c * 32 + o1], &Ks[krow * 72 + c * 32 + o2]);
                    s[nt] = wmma_f16(qf[c], kf, s[nt]);
                }
            }

            // causal mask: only the diagonal tiles need it (scalar branch)
            if (kt + 31 > q0u) {
                #pragma unroll
                for (int nt = 0; nt < 2; ++nt) {
                    const int kj = kt + nt * 16 + hl;   // this lane's key column
                    #pragma unroll
                    for (int r = 0; r < 8; ++r)
                        s[nt][r] = (kj > qlane + r) ? -3e38f : s[nt][r];
                }
            }

            // row max across 16 lanes (xor masks 1..8 stay in each 16-lane half)
            float sm[8];
            #pragma unroll
            for (int r = 0; r < 8; ++r) sm[r] = fmaxf(s[0][r], s[1][r]);
            #pragma unroll
            for (int mk = 1; mk <= 8; mk <<= 1)
                #pragma unroll
                for (int r = 0; r < 8; ++r) sm[r] = fmaxf(sm[r], __shfl_xor(sm[r], mk, 32));

            // online softmax update
            float rs[8];
            #pragma unroll
            for (int r = 0; r < 8; ++r) {
                const float mn = fmaxf(mrow[r], sm[r]);
                const float sc = __expf(mrow[r] - mn);
                mrow[r] = mn;
                const float p0 = __expf(s[0][r] - mn);
                const float p1 = __expf(s[1][r] - mn);
                s[0][r] = p0; s[1][r] = p1;
                rs[r] = p0 + p1;
                lrow[r] *= sc;
                #pragma unroll
                for (int dt = 0; dt < 4; ++dt) oacc[dt][r] *= sc;
            }
            #pragma unroll
            for (int mk = 1; mk <= 8; mk <<= 1)
                #pragma unroll
                for (int r = 0; r < 8; ++r) rs[r] += __shfl_xor(rs[r], mk, 32);
            #pragma unroll
            for (int r = 0; r < 8; ++r) lrow[r] += rs[r];

            // transpose P (D-layout -> A-layout) via wave-private LDS (DS in-order)
            #pragma unroll
            for (int nt = 0; nt < 2; ++nt)
                #pragma unroll
                for (int r = 0; r < 8; ++r)
                    Ps[wv][(r + grp * 8) * 40 + nt * 16 + hl] = (_Float16)s[nt][r];
            v16h pf;
            { const _Float16* pr = &Ps[wv][hl * 40]; pf = ld_frag(pr + o1, pr + o2); }

            // O += P @ V : 4 d-tiles of 16 columns
            #pragma unroll
            for (int dt = 0; dt < 4; ++dt) {
                const int dcol = dt * 16 + hl;
                v16h vf = ld_frag(&Vt[dcol * 40 + o1], &Vt[dcol * 40 + o2]);
                oacc[dt] = wmma_f16(pf, vf, oacc[dt]);
            }
        }

        __syncthreads();   // all waves done reading LDS tile t
        if (t + 1 < nkt) { // commit prefetched tile t+1
            *(v8h*)&Ks[skr * 72 + skc] = kR;
            #pragma unroll
            for (int i = 0; i < 8; ++i) Vt[(svd + i) * 40 + svk] = vR[i];
        }
    }

    // epilogue: normalize and write f16 O, layout [b*N + n][h*64 + d]
    const int bb = bh >> 4, hh = bh & 15;
    float linv[8];
    #pragma unroll
    for (int r = 0; r < 8; ++r) linv[r] = 1.0f / lrow[r];
    #pragma unroll
    for (int dt = 0; dt < 4; ++dt) {
        const int d = dt * 16 + hl;
        #pragma unroll
        for (int r = 0; r < 8; ++r) {
            const int qi = qlane + r;
            O[((size_t)(bb * SEQ + qi)) * DIMM + hh * DHEAD + d] = (_Float16)(oacc[dt][r] * linv[r]);
        }
    }
}

// ---------------------------------------------------------------------------
// Host launcher
// ---------------------------------------------------------------------------
extern "C" void kernel_launch(void* const* d_in, const int* in_sizes, int n_in,
                              void* d_out, int out_size, void* d_ws, size_t ws_size,
                              hipStream_t stream) {
    const float* x    = (const float*)d_in[0];
    const float* lnw  = (const float*)d_in[1];
    const float* lnb  = (const float*)d_in[2];
    const float* wqkv = (const float*)d_in[3];
    const float* wout = (const float*)d_in[4];
    float* out = (float*)d_out;

    // workspace carve (f16 buffers)
    char* p = (char*)d_ws;
    _Float16* xn     = (_Float16*)p; p += (size_t)BROWS * DIMM * 2;          // 8 MB (reused as O)
    _Float16* wqkvh  = (_Float16*)p; p += (size_t)DIMM * 3 * DIMM * 2;       // 6 MB
    _Float16* wouth  = (_Float16*)p; p += (size_t)DIMM * DIMM * 2;           // 2 MB
    _Float16* qb     = (_Float16*)p; p += (size_t)BROWS * DIMM * 2;          // 8 MB
    _Float16* kb     = (_Float16*)p; p += (size_t)BROWS * DIMM * 2;          // 8 MB
    _Float16* vb     = (_Float16*)p; p += (size_t)BROWS * DIMM * 2;          // 8 MB
    _Float16* ob     = xn;  // xn is dead after QKV GEMM; alias for O

    // 1) LayerNorm -> f16
    ln_kernel<<<BROWS, 256, 0, stream>>>(x, lnw, lnb, xn);
    // 2) weight casts (4 elems/thread)
    cast_f16_kernel<<<(DIMM * 3 * DIMM / 4 + 255) / 256, 256, 0, stream>>>(wqkv, wqkvh, DIMM * 3 * DIMM / 4);
    cast_f16_kernel<<<(DIMM * DIMM / 4 + 255) / 256, 256, 0, stream>>>(wout, wouth, DIMM * DIMM / 4);
    // 3) QKV projection -> head-major q (scaled), k, v
    gemm_wmma<0><<<dim3(3 * DIMM / 128, BROWS / 128), 256, 0, stream>>>(
        xn, wqkvh, BROWS, 3 * DIMM, DIMM, qb, kb, vb, nullptr);
    // 4) causal flash attention (128 query rows / block)
    attn_kernel<<<dim3(2 * NHEAD, SEQ / 128), 256, 0, stream>>>(qb, kb, vb, ob);
    // 5) output projection -> fp32
    gemm_wmma<1><<<dim3(DIMM / 128, BROWS / 128), 256, 0, stream>>>(
        ob, wouth, BROWS, DIMM, DIMM, nullptr, nullptr, nullptr, out);
}
